// EGTF_77653008712091
// MI455X (gfx1250) — compile-verified, matrix operand order checked by reading, not run on hardware
//
#include <hip/hip_runtime.h>
#include <stdint.h>

#define N_ATOMS 4096
#define N_EDGES 131072
#define HDIM    256
#define NLAYERS 4
#define NGRAPH  64
#define NEUR    512
#define NHEADS  8
#define DHEAD   32

typedef __bf16 bf16_t;
typedef __attribute__((ext_vector_type(16))) __bf16 bf16x16;
typedef __attribute__((ext_vector_type(8)))  __bf16 bf16x8;
typedef __attribute__((ext_vector_type(8)))  float  f32x8;

__device__ __forceinline__ f32x8 zero8() {
  f32x8 v = {0.f,0.f,0.f,0.f,0.f,0.f,0.f,0.f};
  return v;
}

// Per-lane A/B fragment for v_wmma_f32_16x16x32_bf16.
// ISA 7.12.2 (16-bit 16x32): lane half h: chunk0 = K[8h..8h+8), chunk1 = K[16+8h..16+8h+8)
__device__ __forceinline__ bf16x16 load_frag(const bf16_t* p, int kb, int half) {
  union { bf16x16 f; bf16x8 h[2]; } u;
  u.h[0] = *(const bf16x8*)(p + kb + (half << 3));
  u.h[1] = *(const bf16x8*)(p + kb + 16 + (half << 3));
  return u.f;
}

__device__ __forceinline__ f32x8 wmma_bf16(bf16x16 a, bf16x16 b, f32x8 c) {
  return __builtin_amdgcn_wmma_f32_16x16x32_bf16(false, a, false, b, (short)0, c, false, false);
}

__device__ __forceinline__ float redsum16(float x) {
  #pragma unroll
  for (int m = 1; m < 16; m <<= 1) x += __shfl_xor(x, m, 32);
  return x;
}
__device__ __forceinline__ float redmax16(float x) {
  #pragma unroll
  for (int m = 1; m < 16; m <<= 1) x = fmaxf(x, __shfl_xor(x, m, 32));
  return x;
}

__device__ __forceinline__ float sigmoid_f(float x) { return 1.f / (1.f + __expf(-x)); }
__device__ __forceinline__ float silu_f(float x)    { return x * sigmoid_f(x); }

__device__ __forceinline__ void atomic_add_f32(float* p, float v) {
  __hip_atomic_fetch_add(p, v, __ATOMIC_RELAXED, __HIP_MEMORY_SCOPE_AGENT);
}

// ---- async weight staging: copy one 256(N) x 32(K) bf16 slice (16 KB) to LDS ----
// 128 threads x 8 b128 chunks = 1024 x 16B.  Generic LDS pointer low 32 bits = LDS offset.
__device__ __forceinline__ void stage_slice(bf16_t* ldsDst, const bf16_t* Wt, int kb, int tid) {
  #pragma unroll
  for (int t = 0; t < 8; ++t) {
    int c = tid + t * 128;                 // b128 chunk id 0..1023
    int n = c >> 2, inner = c & 3;
    const bf16_t* g = Wt + (size_t)n * HDIM + kb + inner * 8;
    uint32_t lo = (uint32_t)(uintptr_t)(ldsDst + (size_t)c * 8);
    asm volatile("global_load_async_to_lds_b128 %0, %1, off" :: "v"(lo), "v"(g) : "memory");
  }
}

// GEMM over NS K=256 sources into a 16(M) x 256(N) strip per wave.
// Weights double-buffered in LDS via async copies; all 4 waves share each slice.
// B fragments software-pipelined (bcur/bnext); A fragment for the next k-step is
// prefetched under the current step's WMMAs.
template <int NS>
__device__ __forceinline__ void gemm_staged(
    f32x8 acc[16],
    const bf16_t* a0, const bf16_t* w0,
    const bf16_t* a1, const bf16_t* w1,
    bf16_t* ldsB,                 // 2 * 8192 bf16 (2 x 16 KB)
    int tid, int half, int l16) {
  const int T = NS * 8;
  stage_slice(ldsB, w0, 0, tid);
  bf16x16 a = load_frag(a0, 0, half);
  #pragma unroll 1
  for (int s = 0; s < T; ++s) {
    int buf = s & 1;
    if (s + 1 < T) {
      const bf16_t* Wn = ((s + 1) < 8) ? w0 : w1;
      stage_slice(ldsB + (size_t)(buf ^ 1) * 8192, Wn, ((s + 1) & 7) * 32, tid);
      asm volatile("s_wait_asynccnt 8" ::: "memory");
    } else {
      asm volatile("s_wait_asynccnt 0" ::: "memory");
    }
    __syncthreads();               // all waves' slice s resident
    bf16x16 anext;
    if (s + 1 < T) {               // prefetch next A frag (covered by WMMAs below)
      const bf16_t* aRowN = (NS == 1 || (s + 1) < 8) ? a0 : a1;
      anext = load_frag(aRowN, ((s + 1) & 7) * 32, half);
    }
    const bf16_t* bbase = ldsB + (size_t)buf * 8192;
    bf16x16 bcur = load_frag(bbase + (size_t)l16 * 32, 0, half);
    #pragma unroll
    for (int nt = 0; nt < 16; ++nt) {
      bf16x16 bnext;
      if (nt < 15)
        bnext = load_frag(bbase + (size_t)(((nt + 1) << 4) + l16) * 32, 0, half);
      acc[nt] = wmma_bf16(a, bcur, acc[nt]);
      bcur = bnext;
    }
    if (s + 1 < T) a = anext;
    __syncthreads();               // slice may be overwritten next iter
  }
}

// ---------------- small utility kernels ----------------

__global__ void k_transpose_cvt(const float* __restrict__ src, bf16_t* __restrict__ dst,
                                int K, int Nc) {
  int idx = blockIdx.x * blockDim.x + threadIdx.x;
  if (idx >= K * Nc) return;
  int k = idx / Nc, n = idx - k * Nc;
  dst[(size_t)n * K + k] = (bf16_t)src[idx];
}

__global__ void k_embed(const int* __restrict__ z, const float* __restrict__ emb,
                        float* __restrict__ h, bf16_t* __restrict__ hb) {
  int idx = blockIdx.x * blockDim.x + threadIdx.x;    // N*H
  int n = idx >> 8, c = idx & 255;
  float v = emb[z[n] * HDIM + c];
  h[idx] = v;
  hb[idx] = (bf16_t)v;
}

__global__ void k_radial(const float* __restrict__ pos, const int* __restrict__ row,
                         const int* __restrict__ col, float* __restrict__ radial) {
  int e = blockIdx.x * blockDim.x + threadIdx.x;
  if (e >= N_EDGES) return;
  int r = row[e], c = col[e];
  float dx = pos[r * 3 + 0] - pos[c * 3 + 0];
  float dy = pos[r * 3 + 1] - pos[c * 3 + 1];
  float dz = pos[r * 3 + 2] - pos[c * 3 + 2];
  radial[e] = dx * dx + dy * dy + dz * dz;
}

__global__ void k_f32_to_bf16(const float* __restrict__ s, bf16_t* __restrict__ d, int n) {
  int idx = blockIdx.x * blockDim.x + threadIdx.x;
  if (idx < n) d[idx] = (bf16_t)s[idx];
}

// ---------------- fused EGNN edge kernel (64 edges / block, 4 waves) ----------------
// m  = silu([h[row]|h[col]|radial] @ eW1 + b1)
// m2 = silu(m @ eW2 + b2);  att = sigmoid(m2 @ aW + ab);  agg[row] += m2*att
__global__ __launch_bounds__(128) void k_edge(
    const bf16_t* __restrict__ hb, const float* __restrict__ radial,
    const int* __restrict__ rowIdx, const int* __restrict__ colIdx,
    const bf16_t* __restrict__ W1a, const bf16_t* __restrict__ W1b,
    const float* __restrict__ wlast, const float* __restrict__ b1,
    const bf16_t* __restrict__ W2t, const float* __restrict__ b2,
    const float* __restrict__ aW, const float* __restrict__ abp,
    float* __restrict__ agg) {
  __shared__ bf16_t ldsB[2 * 8192];      // 32 KB weight staging
  __shared__ bf16_t ldsM[64 * HDIM];     // 32 KB intermediate tile
  int tid = threadIdx.x;
  int wave = tid >> 5, lane = tid & 31;
  int half = lane >> 4, l16 = lane & 15;
  int e0 = blockIdx.x * 64 + wave * 16;

  f32x8 acc[16];
  #pragma unroll
  for (int nt = 0; nt < 16; ++nt) acc[nt] = zero8();

  int eA = e0 + l16;
  const bf16_t* aR = hb + (size_t)rowIdx[eA] * HDIM;
  const bf16_t* aC = hb + (size_t)colIdx[eA] * HDIM;
  gemm_staged<2>(acc, aR, W1a, aC, W1b, ldsB, tid, half, l16);

  float rad[8];
  #pragma unroll
  for (int v = 0; v < 8; ++v) rad[v] = radial[e0 + 8 * half + v];

  #pragma unroll
  for (int nt = 0; nt < 16; ++nt) {
    int n = (nt << 4) + l16;
    float wl = wlast[n], bb = b1[n];
    #pragma unroll
    for (int v = 0; v < 8; ++v) {
      float x = acc[nt][v] + rad[v] * wl + bb;
      ldsM[(wave * 16 + 8 * half + v) * HDIM + n] = (bf16_t)silu_f(x);
    }
  }
  asm volatile("s_wait_dscnt 0" ::: "memory");

  #pragma unroll
  for (int nt = 0; nt < 16; ++nt) acc[nt] = zero8();
  gemm_staged<1>(acc, ldsM + (size_t)(wave * 16 + l16) * HDIM, W2t,
                 (const bf16_t*)0, (const bf16_t*)0, ldsB, tid, half, l16);

  float dot[8];
  #pragma unroll
  for (int v = 0; v < 8; ++v) dot[v] = 0.f;
  #pragma unroll
  for (int nt = 0; nt < 16; ++nt) {
    int n = (nt << 4) + l16;
    float bb = b2[n], aw = aW[n];
    #pragma unroll
    for (int v = 0; v < 8; ++v) {
      float x = silu_f(acc[nt][v] + bb);
      acc[nt][v] = x;
      dot[v] += x * aw;
    }
  }
  float abv = abp[0];
  float att[8];
  #pragma unroll
  for (int v = 0; v < 8; ++v) att[v] = sigmoid_f(redsum16(dot[v]) + abv);

  #pragma unroll
  for (int v = 0; v < 8; ++v) {
    int e = e0 + 8 * half + v;
    float* dst = agg + (size_t)rowIdx[e] * HDIM;
    #pragma unroll
    for (int nt = 0; nt < 16; ++nt)
      atomic_add_f32(&dst[(nt << 4) + l16], acc[nt][v] * att[v]);
  }
}

// ---------------- fused node update (64 nodes / block) ----------------
// o = silu([h|agg] @ nW1 + b1);  h += o @ nW2 + b2
__global__ __launch_bounds__(128) void k_node(
    float* __restrict__ hf, bf16_t* __restrict__ hb, const bf16_t* __restrict__ aggb,
    const bf16_t* __restrict__ W1a, const bf16_t* __restrict__ W1b,
    const float* __restrict__ b1, const bf16_t* __restrict__ W2t,
    const float* __restrict__ b2) {
  __shared__ bf16_t ldsB[2 * 8192];
  __shared__ bf16_t ldsM[64 * HDIM];
  int tid = threadIdx.x;
  int wave = tid >> 5, lane = tid & 31;
  int half = lane >> 4, l16 = lane & 15;
  int n0 = blockIdx.x * 64 + wave * 16;

  f32x8 acc[16];
  #pragma unroll
  for (int nt = 0; nt < 16; ++nt) acc[nt] = zero8();
  gemm_staged<2>(acc, hb + (size_t)(n0 + l16) * HDIM, W1a,
                 aggb + (size_t)(n0 + l16) * HDIM, W1b, ldsB, tid, half, l16);

  #pragma unroll
  for (int nt = 0; nt < 16; ++nt) {
    int n = (nt << 4) + l16;
    float bb = b1[n];
    #pragma unroll
    for (int v = 0; v < 8; ++v)
      ldsM[(wave * 16 + 8 * half + v) * HDIM + n] = (bf16_t)silu_f(acc[nt][v] + bb);
  }
  asm volatile("s_wait_dscnt 0" ::: "memory");

  #pragma unroll
  for (int nt = 0; nt < 16; ++nt) acc[nt] = zero8();
  gemm_staged<1>(acc, ldsM + (size_t)(wave * 16 + l16) * HDIM, W2t,
                 (const bf16_t*)0, (const bf16_t*)0, ldsB, tid, half, l16);

  #pragma unroll
  for (int nt = 0; nt < 16; ++nt) {
    int n = (nt << 4) + l16;
    float bb = b2[n];
    #pragma unroll
    for (int v = 0; v < 8; ++v) {
      size_t idx = (size_t)(n0 + 8 * half + v) * HDIM + n;
      float x = hf[idx] + acc[nt][v] + bb;
      hf[idx] = x;
      hb[idx] = (bf16_t)x;
    }
  }
}

// ---------------- GEMM + bias, store bf16 (optionally transposed) ----------------
__global__ __launch_bounds__(128) void k_gemm_store(
    const bf16_t* __restrict__ A, const bf16_t* __restrict__ Wt,
    const float* __restrict__ bias, bf16_t* __restrict__ out, int transposed) {
  __shared__ bf16_t ldsB[2 * 8192];
  int tid = threadIdx.x;
  int wave = tid >> 5, lane = tid & 31;
  int half = lane >> 4, l16 = lane & 15;
  int n0 = blockIdx.x * 64 + wave * 16;

  f32x8 acc[16];
  #pragma unroll
  for (int nt = 0; nt < 16; ++nt) acc[nt] = zero8();
  gemm_staged<1>(acc, A + (size_t)(n0 + l16) * HDIM, Wt,
                 (const bf16_t*)0, (const bf16_t*)0, ldsB, tid, half, l16);

  #pragma unroll
  for (int nt = 0; nt < 16; ++nt) {
    int n = (nt << 4) + l16;
    float bb = bias[n];
    #pragma unroll
    for (int v = 0; v < 8; ++v) {
      float x = acc[nt][v] + bb;
      int node = n0 + 8 * half + v;
      if (transposed) out[(size_t)n * N_ATOMS + node] = (bf16_t)x;
      else            out[(size_t)node * HDIM + n]   = (bf16_t)x;
    }
  }
}

// ---------------- flash attention: one wave per (16-query tile, head) ----------------
__global__ __launch_bounds__(32) void k_attn(
    const bf16_t* __restrict__ q, const bf16_t* __restrict__ k,
    const bf16_t* __restrict__ vt, bf16_t* __restrict__ outb) {
  __shared__ bf16_t ldsP[16 * 32];
  int lane = threadIdx.x, half = lane >> 4, l16 = lane & 15;
  int qt = blockIdx.x, head = blockIdx.y;
  int q0 = qt * 16;
  const float scale = 0.17677669529663687f;   // 1/sqrt(32)

  bf16x16 afq = load_frag(q + (size_t)(q0 + l16) * HDIM + head * DHEAD, 0, half);

  f32x8 accO[2];
  accO[0] = zero8(); accO[1] = zero8();
  float rmax[8], rsum[8];
  #pragma unroll
  for (int v = 0; v < 8; ++v) { rmax[v] = -3.0e38f; rsum[v] = 0.f; }

  // prime first key-tile fragments
  bf16x16 kf0 = load_frag(k + (size_t)(l16) * HDIM + head * DHEAD, 0, half);
  bf16x16 kf1 = load_frag(k + (size_t)(16 + l16) * HDIM + head * DHEAD, 0, half);

  #pragma unroll 1
  for (int kb0 = 0; kb0 < N_ATOMS; kb0 += 32) {
    // V fragments for this tile issued early; consumed only after softmax
    bf16x16 vf0 = load_frag(vt + (size_t)(head * DHEAD + l16) * N_ATOMS + kb0, 0, half);
    bf16x16 vf1 = load_frag(vt + (size_t)(head * DHEAD + 16 + l16) * N_ATOMS + kb0, 0, half);

    f32x8 s0 = wmma_bf16(afq, kf0, zero8());
    f32x8 s1 = wmma_bf16(afq, kf1, zero8());

    if (kb0 + 32 < N_ATOMS) {   // prefetch next tile's K fragments under softmax
      kf0 = load_frag(k + (size_t)(kb0 + 32 + l16) * HDIM + head * DHEAD, 0, half);
      kf1 = load_frag(k + (size_t)(kb0 + 48 + l16) * HDIM + head * DHEAD, 0, half);
    }

    #pragma unroll
    for (int v = 0; v < 8; ++v) {
      float sv0 = s0[v] * scale, sv1 = s1[v] * scale;
      float mv = redmax16(fmaxf(sv0, sv1));
      float nmax = fmaxf(rmax[v], mv);
      float corr = __expf(rmax[v] - nmax);
      rmax[v] = nmax;
      float p0 = __expf(sv0 - nmax);
      float p1 = __expf(sv1 - nmax);
      ldsP[(8 * half + v) * 32 + l16]      = (bf16_t)p0;
      ldsP[(8 * half + v) * 32 + 16 + l16] = (bf16_t)p1;
      rsum[v] = rsum[v] * corr + redsum16(p0 + p1);
      accO[0][v] *= corr;
      accO[1][v] *= corr;
    }
    asm volatile("s_wait_dscnt 0" ::: "memory");
    bf16x16 pf = load_frag(ldsP + (size_t)l16 * 32, 0, half);
    accO[0] = wmma_bf16(pf, vf0, accO[0]);
    accO[1] = wmma_bf16(pf, vf1, accO[1]);
  }

  #pragma unroll
  for (int t2 = 0; t2 < 2; ++t2)
    #pragma unroll
    for (int v = 0; v < 8; ++v) {
      int node = q0 + 8 * half + v;
      outb[(size_t)node * HDIM + head * DHEAD + t2 * 16 + l16] =
          (bf16_t)(accO[t2][v] / rsum[v]);
    }
}

// ---------------- GEMM + bias + residual + LayerNorm ----------------
__global__ __launch_bounds__(128) void k_gemm_res_ln(
    const bf16_t* __restrict__ A, const bf16_t* __restrict__ Wt,
    const float* __restrict__ bias, float* __restrict__ hf, bf16_t* __restrict__ hb,
    const float* __restrict__ lng, const float* __restrict__ lnb) {
  __shared__ bf16_t ldsB[2 * 8192];
  int tid = threadIdx.x;
  int wave = tid >> 5, lane = tid & 31;
  int half = lane >> 4, l16 = lane & 15;
  int n0 = blockIdx.x * 64 + wave * 16;

  f32x8 acc[16];
  #pragma unroll
  for (int nt = 0; nt < 16; ++nt) acc[nt] = zero8();
  gemm_staged<1>(acc, A + (size_t)(n0 + l16) * HDIM, Wt,
                 (const bf16_t*)0, (const bf16_t*)0, ldsB, tid, half, l16);

  float s[8], ss[8];
  #pragma unroll
  for (int v = 0; v < 8; ++v) { s[v] = 0.f; ss[v] = 0.f; }
  #pragma unroll
  for (int nt = 0; nt < 16; ++nt) {
    int n = (nt << 4) + l16;
    float bb = bias[n];
    #pragma unroll
    for (int v = 0; v < 8; ++v) {
      float x = acc[nt][v] + bb + hf[(size_t)(n0 + 8 * half + v) * HDIM + n];
      acc[nt][v] = x;
      s[v] += x; ss[v] += x * x;
    }
  }
  #pragma unroll
  for (int v = 0; v < 8; ++v) {
    float mean = redsum16(s[v]) * (1.f / HDIM);
    float var  = redsum16(ss[v]) * (1.f / HDIM) - mean * mean;
    s[v] = mean; ss[v] = rsqrtf(var + 1e-5f);
  }
  #pragma unroll
  for (int nt = 0; nt < 16; ++nt) {
    int n = (nt << 4) + l16;
    float g = lng[n], b = lnb[n];
    #pragma unroll
    for (int v = 0; v < 8; ++v) {
      size_t idx = (size_t)(n0 + 8 * half + v) * HDIM + n;
      float y = (acc[nt][v] - s[v]) * ss[v] * g + b;
      hf[idx] = y;
      hb[idx] = (bf16_t)y;
    }
  }
}

// ---------------- fused FFN: silu(h@W1+b1)@W2+b2, residual + LN ----------------
__global__ __launch_bounds__(128) void k_ffn(
    float* __restrict__ hf, bf16_t* __restrict__ hb,
    const bf16_t* __restrict__ W1t, const float* __restrict__ b1,
    const bf16_t* __restrict__ W2t, const float* __restrict__ b2,
    const float* __restrict__ lng, const float* __restrict__ lnb) {
  __shared__ bf16_t ldsB[2 * 8192];
  __shared__ bf16_t ldsM[64 * HDIM];
  int tid = threadIdx.x;
  int wave = tid >> 5, lane = tid & 31;
  int half = lane >> 4, l16 = lane & 15;
  int n0 = blockIdx.x * 64 + wave * 16;

  f32x8 acc[16];
  #pragma unroll
  for (int nt = 0; nt < 16; ++nt) acc[nt] = zero8();
  gemm_staged<1>(acc, hb + (size_t)(n0 + l16) * HDIM, W1t,
                 (const bf16_t*)0, (const bf16_t*)0, ldsB, tid, half, l16);

  #pragma unroll
  for (int nt = 0; nt < 16; ++nt) {
    int n = (nt << 4) + l16;
    float bb = b1[n];
    #pragma unroll
    for (int v = 0; v < 8; ++v)
      ldsM[(wave * 16 + 8 * half + v) * HDIM + n] = (bf16_t)silu_f(acc[nt][v] + bb);
  }
  asm volatile("s_wait_dscnt 0" ::: "memory");

  #pragma unroll
  for (int nt = 0; nt < 16; ++nt) acc[nt] = zero8();
  gemm_staged<1>(acc, ldsM + (size_t)(wave * 16 + l16) * HDIM, W2t,
                 (const bf16_t*)0, (const bf16_t*)0, ldsB, tid, half, l16);

  float s[8], ss[8];
  #pragma unroll
  for (int v = 0; v < 8; ++v) { s[v] = 0.f; ss[v] = 0.f; }
  #pragma unroll
  for (int nt = 0; nt < 16; ++nt) {
    int n = (nt << 4) + l16;
    float bb = b2[n];
    #pragma unroll
    for (int v = 0; v < 8; ++v) {
      float x = acc[nt][v] + bb + hf[(size_t)(n0 + 8 * half + v) * HDIM + n];
      acc[nt][v] = x;
      s[v] += x; ss[v] += x * x;
    }
  }
  #pragma unroll
  for (int v = 0; v < 8; ++v) {
    float mean = redsum16(s[v]) * (1.f / HDIM);
    float var  = redsum16(ss[v]) * (1.f / HDIM) - mean * mean;
    s[v] = mean; ss[v] = rsqrtf(var + 1e-5f);
  }
  #pragma unroll
  for (int nt = 0; nt < 16; ++nt) {
    int n = (nt << 4) + l16;
    float g = lng[n], b = lnb[n];
    #pragma unroll
    for (int v = 0; v < 8; ++v) {
      size_t idx = (size_t)(n0 + 8 * half + v) * HDIM + n;
      float y = (acc[nt][v] - s[v]) * ss[v] * g + b;
      hf[idx] = y;
      hb[idx] = (bf16_t)y;
    }
  }
}

// ---------------- pool + energy head ----------------
__global__ void k_pool(const float* __restrict__ hf, const int* __restrict__ batch,
                       float* __restrict__ hg) {
  int idx = blockIdx.x * blockDim.x + threadIdx.x;   // N*H
  int n = idx >> 8, c = idx & 255;
  atomic_add_f32(&hg[(size_t)batch[n] * HDIM + c], hf[idx]);
}

__global__ __launch_bounds__(512) void k_head(
    const float* __restrict__ hg, const float* __restrict__ W1, const float* __restrict__ b1,
    const float* __restrict__ W2, const float* __restrict__ b2, float* __restrict__ out) {
  __shared__ float red[NEUR];
  int g = blockIdx.x, j = threadIdx.x;
  float acc = b1[j];
  for (int c = 0; c < HDIM; ++c) acc += hg[(size_t)g * HDIM + c] * W1[(size_t)c * NEUR + j];
  red[j] = silu_f(acc) * W2[j];
  __syncthreads();
  for (int st = NEUR / 2; st > 0; st >>= 1) {
    if (j < st) red[j] += red[j + st];
    __syncthreads();
  }
  if (j == 0) out[g] = red[0] + b2[0];
}

// ---------------- host launch ----------------
extern "C" void kernel_launch(void* const* d_in, const int* in_sizes, int n_in,
                              void* d_out, int out_size, void* d_ws, size_t ws_size,
                              hipStream_t stream) {
  const int*   z    = (const int*)d_in[0];
  const float* pos  = (const float*)d_in[1];
  const int*   bat  = (const int*)d_in[2];
  const int*   eidx = (const int*)d_in[3];
  const float* emb  = (const float*)d_in[4];
  const float* eW1  = (const float*)d_in[5];
  const float* eb1  = (const float*)d_in[6];
  const float* eW2  = (const float*)d_in[7];
  const float* eb2  = (const float*)d_in[8];
  const float* aW   = (const float*)d_in[9];
  const float* ab   = (const float*)d_in[10];
  const float* nW1  = (const float*)d_in[11];
  const float* nb1  = (const float*)d_in[12];
  const float* nW2  = (const float*)d_in[13];
  const float* nb2  = (const float*)d_in[14];
  const float* Wq   = (const float*)d_in[15];
  const float* bq   = (const float*)d_in[16];
  const float* Wk   = (const float*)d_in[17];
  const float* bk   = (const float*)d_in[18];
  const float* Wv   = (const float*)d_in[19];
  const float* bv   = (const float*)d_in[20];
  const float* Wo   = (const float*)d_in[21];
  const float* bo   = (const float*)d_in[22];
  const float* ln1g = (const float*)d_in[23];
  const float* ln1b = (const float*)d_in[24];
  const float* fW1  = (const float*)d_in[25];
  const float* fb1  = (const float*)d_in[26];
  const float* fW2  = (const float*)d_in[27];
  const float* fb2  = (const float*)d_in[28];
  const float* ln2g = (const float*)d_in[29];
  const float* ln2b = (const float*)d_in[30];
  const float* gW1  = (const float*)d_in[31];
  const float* gb1  = (const float*)d_in[32];
  const float* gW2  = (const float*)d_in[33];
  const float* gb2  = (const float*)d_in[34];

  const int* rowI = eidx;
  const int* colI = eidx + N_EDGES;

  char* ws = (char*)d_ws;
  size_t off = 0;
  auto alloc = [&](size_t bytes) -> char* {
    char* p = ws + off;
    off += (bytes + 255) & ~(size_t)255;
    return p;
  };

  const size_t NH = (size_t)N_ATOMS * HDIM;
  float*  hf   = (float*)alloc(NH * 4);
  bf16_t* hb   = (bf16_t*)alloc(NH * 2);
  float*  rad  = (float*)alloc((size_t)N_EDGES * 4);
  float*  aggf = (float*)alloc(NH * 4);
  bf16_t* aggb = (bf16_t*)alloc(NH * 2);
  bf16_t* qb   = (bf16_t*)alloc(NH * 2);
  bf16_t* kb   = (bf16_t*)alloc(NH * 2);
  bf16_t* vtb  = (bf16_t*)alloc(NH * 2);
  bf16_t* atb  = (bf16_t*)alloc(NH * 2);
  float*  hg   = (float*)alloc((size_t)NGRAPH * HDIM * 4);

  const size_t WSZ = (size_t)HDIM * HDIM;   // 65536 elements
  bf16_t *W1aT[NLAYERS], *W1bT[NLAYERS], *eW2T[NLAYERS];
  bf16_t *nW1aT[NLAYERS], *nW1bT[NLAYERS], *nW2T[NLAYERS];
  for (int l = 0; l < NLAYERS; ++l) {
    W1aT[l]  = (bf16_t*)alloc(WSZ * 2);
    W1bT[l]  = (bf16_t*)alloc(WSZ * 2);
    eW2T[l]  = (bf16_t*)alloc(WSZ * 2);
    nW1aT[l] = (bf16_t*)alloc(WSZ * 2);
    nW1bT[l] = (bf16_t*)alloc(WSZ * 2);
    nW2T[l]  = (bf16_t*)alloc(WSZ * 2);
  }
  bf16_t* WqT  = (bf16_t*)alloc(WSZ * 2);
  bf16_t* WkT  = (bf16_t*)alloc(WSZ * 2);
  bf16_t* WvT  = (bf16_t*)alloc(WSZ * 2);
  bf16_t* WoT  = (bf16_t*)alloc(WSZ * 2);
  bf16_t* fW1T = (bf16_t*)alloc(WSZ * 2);
  bf16_t* fW2T = (bf16_t*)alloc(WSZ * 2);

  dim3 tb(256);
  dim3 tg((unsigned)((WSZ + 255) / 256));

  // weight transpose + bf16 convert
  for (int l = 0; l < NLAYERS; ++l) {
    const float* e1 = eW1 + (size_t)l * 513 * HDIM;
    k_transpose_cvt<<<tg, tb, 0, stream>>>(e1,                 W1aT[l], HDIM, HDIM);
    k_transpose_cvt<<<tg, tb, 0, stream>>>(e1 + WSZ,           W1bT[l], HDIM, HDIM);
    k_transpose_cvt<<<tg, tb, 0, stream>>>(eW2 + (size_t)l * WSZ, eW2T[l], HDIM, HDIM);
    const float* n1 = nW1 + (size_t)l * 512 * HDIM;
    k_transpose_cvt<<<tg, tb, 0, stream>>>(n1,                 nW1aT[l], HDIM, HDIM);
    k_transpose_cvt<<<tg, tb, 0, stream>>>(n1 + WSZ,           nW1bT[l], HDIM, HDIM);
    k_transpose_cvt<<<tg, tb, 0, stream>>>(nW2 + (size_t)l * WSZ, nW2T[l], HDIM, HDIM);
  }
  k_transpose_cvt<<<tg, tb, 0, stream>>>(Wq,  WqT,  HDIM, HDIM);
  k_transpose_cvt<<<tg, tb, 0, stream>>>(Wk,  WkT,  HDIM, HDIM);
  k_transpose_cvt<<<tg, tb, 0, stream>>>(Wv,  WvT,  HDIM, HDIM);
  k_transpose_cvt<<<tg, tb, 0, stream>>>(Wo,  WoT,  HDIM, HDIM);
  k_transpose_cvt<<<tg, tb, 0, stream>>>(fW1, fW1T, HDIM, HDIM);
  k_transpose_cvt<<<tg, tb, 0, stream>>>(fW2, fW2T, HDIM, HDIM);

  // embedding + radial
  k_embed<<<dim3((unsigned)(NH / 256)), tb, 0, stream>>>(z, emb, hf, hb);
  k_radial<<<dim3(N_EDGES / 256), tb, 0, stream>>>(pos, rowI, colI, rad);

  dim3 tb128(128);
  // EGNN layers
  for (int l = 0; l < NLAYERS; ++l) {
    hipMemsetAsync(aggf, 0, NH * 4, stream);
    const float* wlast = eW1 + (size_t)l * 513 * HDIM + 2 * WSZ;  // last concat row (radial)
    k_edge<<<dim3(N_EDGES / 64), tb128, 0, stream>>>(
        hb, rad, rowI, colI,
        W1aT[l], W1bT[l], wlast, eb1 + (size_t)l * HDIM,
        eW2T[l], eb2 + (size_t)l * HDIM,
        aW + (size_t)l * HDIM, ab + l, aggf);
    k_f32_to_bf16<<<dim3((unsigned)(NH / 256)), tb, 0, stream>>>(aggf, aggb, (int)NH);
    k_node<<<dim3(N_ATOMS / 64), tb128, 0, stream>>>(
        hf, hb, aggb,
        nW1aT[l], nW1bT[l], nb1 + (size_t)l * HDIM,
        nW2T[l], nb2 + (size_t)l * HDIM);
  }

  // transformer encoder
  dim3 gN(N_ATOMS / 64);
  k_gemm_store<<<gN, tb128, 0, stream>>>(hb, WqT, bq, qb, 0);
  k_gemm_store<<<gN, tb128, 0, stream>>>(hb, WkT, bk, kb, 0);
  k_gemm_store<<<gN, tb128, 0, stream>>>(hb, WvT, bv, vtb, 1);   // store v transposed [H][N]
  k_attn<<<dim3(N_ATOMS / 16, NHEADS), dim3(32), 0, stream>>>(qb, kb, vtb, atb);
  k_gemm_res_ln<<<gN, tb128, 0, stream>>>(atb, WoT, bo, hf, hb, ln1g, ln1b);
  k_ffn<<<gN, tb128, 0, stream>>>(hf, hb, fW1T, fb1, fW2T, fb2, ln2g, ln2b);

  // pool + head
  hipMemsetAsync(hg, 0, (size_t)NGRAPH * HDIM * 4, stream);
  k_pool<<<dim3((unsigned)(NH / 256)), tb, 0, stream>>>(hf, bat, hg);
  k_head<<<dim3(NGRAPH), dim3(NEUR), 0, stream>>>(hg, gW1, gb1, gW2, gb2, (float*)d_out);
}
